// FusionModel_43267500540827
// MI455X (gfx1250) — matrix operand order repaired and checked by reading, not compiled, not run on hardware
//
#include <hip/hip_runtime.h>
#include <hip/hip_bf16.h>

// ---------------------------------------------------------------------------
// MI455X (gfx1250) implementation. wave32, WMMA f32_16x16x32_f16 everywhere.
// GEMMs are register-blocked 32x32 per wave: 4 independent WMMA accumulators
// per k-step to cover the f16 WMMA->WMMA RAW hazard and double LDS reuse.
// ---------------------------------------------------------------------------

typedef __attribute__((ext_vector_type(16))) _Float16 v16h;
typedef __attribute__((ext_vector_type(8)))  float    v8f;
typedef _Float16 h16;

#define N_NODES 6144
#define D_IN    64
#define GAT_H   64
#define HEADS   8
#define CNN_H   128
#define TR_H    128
#define NH      2
#define HD      64
#define FF_DIM  2048
#define OUT_F   32
#define CNN_FEAT 6144   // D_IN * 96
#define FUSED_F  320    // GAT_H + CNN_H + TR_H

// ---------------------------------------------------------------------------
// Elementwise helpers
// ---------------------------------------------------------------------------
__global__ void fill_f32(float* __restrict__ p, float v, size_t n) {
  size_t i = (size_t)blockIdx.x * blockDim.x + threadIdx.x;
  if (i < n) p[i] = v;
}

__global__ void f32_to_f16(const float* __restrict__ s, h16* __restrict__ d, size_t n) {
  size_t i = (size_t)blockIdx.x * blockDim.x + threadIdx.x;
  if (i < n) d[i] = (h16)s[i];
}

// x[n,f] = act(x[n,f] + bias[f]); optionally write f16 copy at (n*ldh + f)
__global__ void bias_act_store(float* __restrict__ x, const float* __restrict__ bias,
                               h16* __restrict__ xh, int N, int F, int ldh, int relu) {
  size_t i = (size_t)blockIdx.x * blockDim.x + threadIdx.x;
  if (i >= (size_t)N * F) return;
  int n = (int)(i / F), f = (int)(i - (size_t)n * F);
  float v = x[i] + bias[f];
  if (relu) v = fmaxf(v, 0.0f);
  x[i] = v;
  if (xh) xh[(size_t)n * ldh + f] = (h16)v;
}

__global__ void fix_neg_inf(float* __restrict__ m, size_t n) {
  size_t i = (size_t)blockIdx.x * blockDim.x + threadIdx.x;
  if (i < n && m[i] < -1e29f) m[i] = 0.0f;
}

// ---------------------------------------------------------------------------
// Generic f16 WMMA GEMM: C(MxN) = act(A(MxK) @ B(KxN) + bias)
// One wave per 32x32 output tile (2x2 register blocking, 4 accumulators).
// K multiple of 32, M,N multiples of 32.
// ---------------------------------------------------------------------------
__global__ __launch_bounds__(32)
void gemm_wmma_f16(const h16* __restrict__ A, const h16* __restrict__ B,
                   const float* __restrict__ bias,
                   float* __restrict__ Cf, h16* __restrict__ Ch,
                   int M, int N, int K, int lda, int ldb, int ldcf, int ldch,
                   int relu, float hscale) {
  __shared__ h16 As[32 * 32];   // 32 rows x 32 k
  __shared__ h16 Bt[32 * 33];   // transposed B tile: [col][k], padded stride

  const int l  = threadIdx.x;
  const int n0 = blockIdx.x * 32;
  const int m0 = blockIdx.y * 32;
  const int mA = l & 15;        // fragment row / col within 16-subtile
  const int hA = l >> 4;        // half selector
  const int kbA = hA * 8;       // unused (A and B both use kb2)
  const int kb2 = hA * 16;      // B fragment K base

  v8f acc00 = {}, acc01 = {}, acc10 = {}, acc11 = {};

  for (int k0 = 0; k0 < K; k0 += 32) {
    // --- stage A tile: 32 rows x 32 k (each lane: one row = 4x16B) ---
    {
      const h16* src = A + (size_t)(m0 + l) * lda + k0;
      const float4* s4 = reinterpret_cast<const float4*>(src);
      float4* d4 = reinterpret_cast<float4*>(&As[l * 32]);
      d4[0] = s4[0]; d4[1] = s4[1]; d4[2] = s4[2]; d4[3] = s4[3];
    }
    // --- stage B tile transposed: 32 k-rows x 32 cols (lane = one k-row) ---
    {
      const h16* src = B + (size_t)(k0 + l) * ldb + n0;
      alignas(16) h16 tmp[32];
      const float4* s4 = reinterpret_cast<const float4*>(src);
      *reinterpret_cast<float4*>(tmp)      = s4[0];
      *reinterpret_cast<float4*>(tmp + 8)  = s4[1];
      *reinterpret_cast<float4*>(tmp + 16) = s4[2];
      *reinterpret_cast<float4*>(tmp + 24) = s4[3];
#pragma unroll
      for (int j = 0; j < 32; ++j) Bt[j * 33 + l] = tmp[j];
    }
    if (k0 + 32 < K) {
      __builtin_prefetch(A + (size_t)(m0 + l) * lda + k0 + 32, 0, 0);
      __builtin_prefetch(B + (size_t)(k0 + 32 + l) * ldb + n0, 0, 0);
    }
    __syncthreads();

    // --- gather fragments (A and B both in contiguous k = 16*h + s order) ---
    v16h a0, a1, b0, b1;
#pragma unroll
    for (int v = 0; v < 8; ++v) {
      int kk = kb2 + 2 * v;
      a0[2 * v]     = As[mA * 32 + kk];
      a0[2 * v + 1] = As[mA * 32 + kk + 1];
      a1[2 * v]     = As[(16 + mA) * 32 + kk];
      a1[2 * v + 1] = As[(16 + mA) * 32 + kk + 1];
    }
#pragma unroll
    for (int e = 0; e < 16; ++e) {
      b0[e] = Bt[mA * 33 + kb2 + e];
      b1[e] = Bt[(16 + mA) * 33 + kb2 + e];
    }

    // 4 independent accumulation chains -> WMMAs pipeline back-to-back
    acc00 = __builtin_amdgcn_wmma_f32_16x16x32_f16(false, a0, false, b0, (short)0, acc00, false, false);
    acc01 = __builtin_amdgcn_wmma_f32_16x16x32_f16(false, a0, false, b1, (short)0, acc01, false, false);
    acc10 = __builtin_amdgcn_wmma_f32_16x16x32_f16(false, a1, false, b0, (short)0, acc10, false, false);
    acc11 = __builtin_amdgcn_wmma_f32_16x16x32_f16(false, a1, false, b1, (short)0, acc11, false, false);
    __syncthreads();
  }

  // --- epilogue: bias + act + f32/f16 stores ---
  const int nA = n0 + mA, nB = n0 + 16 + mA;
  const float bvA = bias ? bias[nA] : 0.0f;
  const float bvB = bias ? bias[nB] : 0.0f;
#pragma unroll
  for (int r = 0; r < 8; ++r) {
    int mLo = m0 + r + hA * 8;
    int mHi = mLo + 16;
    float v00 = acc00[r] + bvA, v01 = acc01[r] + bvB;
    float v10 = acc10[r] + bvA, v11 = acc11[r] + bvB;
    if (relu) {
      v00 = fmaxf(v00, 0.0f); v01 = fmaxf(v01, 0.0f);
      v10 = fmaxf(v10, 0.0f); v11 = fmaxf(v11, 0.0f);
    }
    if (Cf) {
      Cf[(size_t)mLo * ldcf + nA] = v00; Cf[(size_t)mLo * ldcf + nB] = v01;
      Cf[(size_t)mHi * ldcf + nA] = v10; Cf[(size_t)mHi * ldcf + nB] = v11;
    }
    if (Ch) {
      Ch[(size_t)mLo * ldch + nA] = (h16)(v00 * hscale);
      Ch[(size_t)mLo * ldch + nB] = (h16)(v01 * hscale);
      Ch[(size_t)mHi * ldch + nA] = (h16)(v10 * hscale);
      Ch[(size_t)mHi * ldch + nB] = (h16)(v11 * hscale);
    }
  }
}

// ---------------------------------------------------------------------------
// Flash attention: one wave per (16-row query tile, head). K/V chunks of 32.
// Q is pre-scaled by 1/sqrt(HD). Never materializes the 6144x6144 scores.
// ---------------------------------------------------------------------------
__global__ __launch_bounds__(32)
void flash_attn_wmma(const h16* __restrict__ Qh, const h16* __restrict__ Kh,
                     const h16* __restrict__ Vh, h16* __restrict__ Oh,
                     int Ntok, int ld) {
  __shared__ h16  Qs[16 * 64];
  __shared__ h16  Ks[32 * 64];
  __shared__ h16  Vs[32 * 64];
  __shared__ float Sf[16 * 32];
  __shared__ h16   Ps[16 * 32];
  __shared__ float rowM[16], rowL[16], rowScale[16], rowInv[16];

  const int l     = threadIdx.x;
  const int qt    = blockIdx.x;
  const int head  = blockIdx.y;
  const int cbase = head * HD;
  const int mA = l & 15, hA = l >> 4, kbA = hA * 8, kb2 = hA * 16;

  // load Q tile (16 x 64)
  {
    const h16* src = Qh + (size_t)(qt * 16 + mA) * ld + cbase + hA * 32;
    const float4* s4 = reinterpret_cast<const float4*>(src);
    float4* d4 = reinterpret_cast<float4*>(&Qs[mA * 64 + hA * 32]);
    d4[0] = s4[0]; d4[1] = s4[1];
  }
  if (l < 16) { rowM[l] = -1e30f; rowL[l] = 0.0f; }
  __syncthreads();

  // Q fragments for the two 32-wide k-steps over HD=64 (reused every chunk)
  v16h qa0, qa1;
#pragma unroll
  for (int v = 0; v < 8; ++v) {
    int kk = kb2 + 2 * v;
    qa0[2 * v]     = Qs[mA * 64 + kk];
    qa0[2 * v + 1] = Qs[mA * 64 + kk + 1];
    qa1[2 * v]     = Qs[mA * 64 + 32 + kk];
    qa1[2 * v + 1] = Qs[mA * 64 + 32 + kk + 1];
  }

  v8f o[4] = {v8f{}, v8f{}, v8f{}, v8f{}};

  for (int c0 = 0; c0 < Ntok; c0 += 32) {
    // stage K and V chunk (32 rows x 64 cols each; lane l = one row)
    {
      const h16* ks = Kh + (size_t)(c0 + l) * ld + cbase;
      const h16* vs = Vh + (size_t)(c0 + l) * ld + cbase;
      const float4* k4 = reinterpret_cast<const float4*>(ks);
      const float4* v4 = reinterpret_cast<const float4*>(vs);
      float4* kd = reinterpret_cast<float4*>(&Ks[l * 64]);
      float4* vd = reinterpret_cast<float4*>(&Vs[l * 64]);
      kd[0] = k4[0]; kd[1] = k4[1]; kd[2] = k4[2]; kd[3] = k4[3];
      vd[0] = v4[0]; vd[1] = v4[1]; vd[2] = v4[2]; vd[3] = v4[3];
    }
    __syncthreads();

    // S = Q @ K^T for 32 keys -> two 16x16 tiles, each 2 WMMA (HD=64)
    v8f s0 = {}, s1 = {};
    v16h b;
#pragma unroll
    for (int e = 0; e < 16; ++e) b[e] = Ks[mA * 64 + kb2 + e];
    s0 = __builtin_amdgcn_wmma_f32_16x16x32_f16(false, qa0, false, b, (short)0, s0, false, false);
#pragma unroll
    for (int e = 0; e < 16; ++e) b[e] = Ks[mA * 64 + 32 + kb2 + e];
    s0 = __builtin_amdgcn_wmma_f32_16x16x32_f16(false, qa1, false, b, (short)0, s0, false, false);
#pragma unroll
    for (int e = 0; e < 16; ++e) b[e] = Ks[(16 + mA) * 64 + kb2 + e];
    s1 = __builtin_amdgcn_wmma_f32_16x16x32_f16(false, qa0, false, b, (short)0, s1, false, false);
#pragma unroll
    for (int e = 0; e < 16; ++e) b[e] = Ks[(16 + mA) * 64 + 32 + kb2 + e];
    s1 = __builtin_amdgcn_wmma_f32_16x16x32_f16(false, qa1, false, b, (short)0, s1, false, false);

    // scatter S tiles to LDS (C layout: (m,n) -> vgpr m&7, lane n + 16*(m>>3))
#pragma unroll
    for (int r = 0; r < 8; ++r) {
      int m = r + hA * 8;
      Sf[m * 32 + mA]      = s0[r];
      Sf[m * 32 + 16 + mA] = s1[r];
    }
    __syncthreads();

    // online softmax per row (lanes 0..15 own one row each)
    if (l < 16) {
      float mold = rowM[l], mx = mold;
#pragma unroll 4
      for (int c = 0; c < 32; ++c) mx = fmaxf(mx, Sf[l * 32 + c]);
      float scale = __expf(mold - mx);
      float sum = 0.0f;
#pragma unroll 4
      for (int c = 0; c < 32; ++c) {
        float p = __expf(Sf[l * 32 + c] - mx);
        Ps[l * 32 + c] = (h16)p;
        sum += p;
      }
      rowScale[l] = scale;
      rowM[l] = mx;
      rowL[l] = rowL[l] * scale + sum;
    }
    __syncthreads();

    // rescale output accumulators
#pragma unroll
    for (int r = 0; r < 8; ++r) {
      float sc = rowScale[r + hA * 8];
      o[0][r] *= sc; o[1][r] *= sc; o[2][r] *= sc; o[3][r] *= sc;
    }

    // P(16x32) @ V(32x64): 4 output tiles, one WMMA each (independent accs)
    v16h pa;
#pragma unroll
    for (int v = 0; v < 8; ++v) {
      int kk = kb2 + 2 * v;
      pa[2 * v]     = Ps[mA * 32 + kk];
      pa[2 * v + 1] = Ps[mA * 32 + kk + 1];
    }
#pragma unroll
    for (int t = 0; t < 4; ++t) {
      v16h vb;
#pragma unroll
      for (int e = 0; e < 16; ++e) vb[e] = Vs[(kb2 + e) * 64 + t * 16 + mA];
      o[t] = __builtin_amdgcn_wmma_f32_16x16x32_f16(false, pa, false, vb,
                                                    (short)0, o[t], false, false);
    }
    __syncthreads();
  }

  if (l < 16) rowInv[l] = 1.0f / (rowL[l] + 1e-20f);
  __syncthreads();
#pragma unroll
  for (int t = 0; t < 4; ++t) {
#pragma unroll
    for (int r = 0; r < 8; ++r) {
      int m = r + hA * 8;
      float val = o[t][r] * rowInv[m];
      Oh[(size_t)(qt * 16 + m) * ld + cbase + t * 16 + mA] = (h16)val;
    }
  }
}

// ---------------------------------------------------------------------------
// GAT edge kernels (segment softmax + scatter). ei = [src(E) | dst(E)];
// edges >= E are self-loops.
// ---------------------------------------------------------------------------
__device__ inline void atomicMaxFloat(float* addr, float val) {
  if (val >= 0.0f) atomicMax((int*)addr, __float_as_int(val));
  else             atomicMin((unsigned int*)addr, __float_as_uint(val));
}

__global__ void gat_attn_coef(const float* __restrict__ h,
                              const float* __restrict__ as_,
                              const float* __restrict__ ad_,
                              float* __restrict__ a_s, float* __restrict__ a_d,
                              int N, int heads, int ch) {
  int i = blockIdx.x * blockDim.x + threadIdx.x;
  if (i >= N * heads) return;
  int n = i / heads, hh = i - n * heads;
  const float* hp = h + (size_t)n * heads * ch + hh * ch;
  float ss = 0.0f, dd = 0.0f;
  for (int c = 0; c < ch; ++c) { ss += hp[c] * as_[hh * ch + c]; dd += hp[c] * ad_[hh * ch + c]; }
  a_s[i] = ss; a_d[i] = dd;
}

__global__ void gat_edge_logits(const int* __restrict__ ei, int E, int N, int heads,
                                const float* __restrict__ a_s, const float* __restrict__ a_d,
                                float* __restrict__ ev, float* __restrict__ mbuf) {
  int idx = blockIdx.x * blockDim.x + threadIdx.x;
  int total = (E + N) * heads;
  if (idx >= total) return;
  int e = idx / heads, hh = idx - e * heads;
  int s = (e < E) ? ei[e]     : (e - E);
  int d = (e < E) ? ei[E + e] : (e - E);
  float v = a_s[s * heads + hh] + a_d[d * heads + hh];
  v = (v > 0.0f) ? v : 0.2f * v;            // leaky_relu 0.2
  ev[idx] = v;
  atomicMaxFloat(&mbuf[d * heads + hh], v);
}

__global__ void gat_edge_exp(const int* __restrict__ ei, int E, int N, int heads,
                             const float* __restrict__ mbuf, float* __restrict__ ev,
                             float* __restrict__ den) {
  int idx = blockIdx.x * blockDim.x + threadIdx.x;
  int total = (E + N) * heads;
  if (idx >= total) return;
  int e = idx / heads, hh = idx - e * heads;
  int d = (e < E) ? ei[E + e] : (e - E);
  float ex = __expf(ev[idx] - mbuf[d * heads + hh]);
  ev[idx] = ex;
  atomicAdd(&den[d * heads + hh], ex);
}

__global__ void gat_edge_scatter(const int* __restrict__ ei, int E, int N, int heads, int ch,
                                 const float* __restrict__ hsrc, const float* __restrict__ ev,
                                 const float* __restrict__ den, float* __restrict__ out) {
  int idx = blockIdx.x * blockDim.x + threadIdx.x;
  int total = (E + N) * heads;
  if (idx >= total) return;
  int e = idx / heads, hh = idx - e * heads;
  int s = (e < E) ? ei[e]     : (e - E);
  int d = (e < E) ? ei[E + e] : (e - E);
  float alpha = ev[idx] / (den[d * heads + hh] + 1e-16f);
  const float* hp = hsrc + (size_t)s * heads * ch + hh * ch;
  float* op = out + (size_t)d * heads * ch + hh * ch;
  for (int c = 0; c < ch; ++c) atomicAdd(&op[c], hp[c] * alpha);
}

// ---------------------------------------------------------------------------
// CNN: three small 1D convs (k=1/3/5, 32ch) + ReLU, written as f16 directly
// into the (N x 6144) concat buffer in reshape(n,-1) layout [x1|x2|x3].
// ---------------------------------------------------------------------------
__global__ void cnn_conv_concat(const float* __restrict__ X,
                                const float* __restrict__ w1, const float* __restrict__ b1,
                                const float* __restrict__ w2, const float* __restrict__ b2,
                                const float* __restrict__ w3, const float* __restrict__ b3,
                                h16* __restrict__ concat, int N) {
  size_t idx = (size_t)blockIdx.x * blockDim.x + threadIdx.x;
  if (idx >= (size_t)N * CNN_FEAT) return;
  int n = (int)(idx / CNN_FEAT);
  int f = (int)(idx - (size_t)n * CNN_FEAT);
  int branch = f >> 11;          // /2048
  int r = f & 2047;
  int c = r >> 6;                // channel 0..31
  int p = r & 63;                // position 0..63
  const float* xr = X + (size_t)n * D_IN;
  float y;
  if (branch == 0) {
    y = xr[p] * w1[c] + b1[c];
  } else if (branch == 1) {
    y = b2[c];
#pragma unroll
    for (int t = 0; t < 3; ++t) {
      int q = p - 1 + t;
      if (q >= 0 && q < D_IN) y += xr[q] * w2[c * 3 + t];
    }
  } else {
    y = b3[c];
#pragma unroll
    for (int t = 0; t < 5; ++t) {
      int q = p - 2 + t;
      if (q >= 0 && q < D_IN) y += xr[q] * w3[c * 5 + t];
    }
  }
  concat[idx] = (h16)fmaxf(y, 0.0f);
}

// ---------------------------------------------------------------------------
// LayerNorm over F=128 features; in = a (+ b if given); writes f32/f16.
// ---------------------------------------------------------------------------
__global__ __launch_bounds__(128)
void layernorm_k(const float* __restrict__ a, const float* __restrict__ b,
                 const float* __restrict__ g, const float* __restrict__ be,
                 float* __restrict__ outf, h16* __restrict__ outh, int F) {
  __shared__ float red[128];
  int row = blockIdx.x, t = threadIdx.x;
  float v = a[(size_t)row * F + t] + (b ? b[(size_t)row * F + t] : 0.0f);
  red[t] = v; __syncthreads();
  for (int s = 64; s > 0; s >>= 1) { if (t < s) red[t] += red[t + s]; __syncthreads(); }
  float mu = red[0] / F; __syncthreads();
  float dv = v - mu;
  red[t] = dv * dv; __syncthreads();
  for (int s = 64; s > 0; s >>= 1) { if (t < s) red[t] += red[t + s]; __syncthreads(); }
  float var = red[0] / F;
  float y = dv * rsqrtf(var + 1e-5f) * g[t] + be[t];
  if (outf) outf[(size_t)row * F + t] = y;
  if (outh) outh[(size_t)row * F + t] = (h16)y;
}

// ---------------------------------------------------------------------------
// Host orchestration
// ---------------------------------------------------------------------------
static inline void launch_gemm(hipStream_t s, const h16* A, const h16* B, const float* bias,
                               float* Cf, h16* Ch, int M, int N, int K,
                               int lda, int ldb, int ldcf, int ldch, int relu, float hscale) {
  dim3 grid(N / 32, M / 32);
  gemm_wmma_f16<<<grid, 32, 0, s>>>(A, B, bias, Cf, Ch, M, N, K,
                                    lda, ldb, ldcf, ldch, relu, hscale);
}

extern "C" void kernel_launch(void* const* d_in, const int* in_sizes, int n_in,
                              void* d_out, int out_size, void* d_ws, size_t ws_size,
                              hipStream_t stream) {
  // ---- input mapping (setup_inputs order, params dict flattened in order) ----
  const float* x    = (const float*)d_in[0];
  const float* Xin  = (const float*)d_in[1];
  const float* gW1  = (const float*)d_in[2];
  const float* gas1 = (const float*)d_in[3];
  const float* gad1 = (const float*)d_in[4];
  const float* gb1  = (const float*)d_in[5];
  const float* gW2  = (const float*)d_in[6];
  const float* gas2 = (const float*)d_in[7];
  const float* gad2 = (const float*)d_in[8];
  const float* gb2  = (const float*)d_in[9];
  const float* cw1  = (const float*)d_in[10];
  const float* cb1  = (const float*)d_in[11];
  const float* cw2  = (const float*)d_in[12];
  const float* cb2  = (const float*)d_in[13];
  const float* cw3  = (const float*)d_in[14];
  const float* cb3  = (const float*)d_in[15];
  const float* cfcw = (const float*)d_in[16];
  const float* cfcb = (const float*)d_in[17];
  const float* fiw  = (const float*)d_in[18];
  const float* fib  = (const float*)d_in[19];
  const float* wq   = (const float*)d_in[20];
  const float* bq   = (const float*)d_in[21];
  const float* wk   = (const float*)d_in[22];
  const float* bk   = (const float*)d_in[23];
  const float* wv   = (const float*)d_in[24];
  const float* bv   = (const float*)d_in[25];
  const float* wo   = (const float*)d_in[26];
  const float* bo   = (const float*)d_in[27];
  const float* ln1g = (const float*)d_in[28];
  const float* ln1b = (const float*)d_in[29];
  const float* ff1w = (const float*)d_in[30];
  const float* ff1b = (const float*)d_in[31];
  const float* ff2w = (const float*)d_in[32];
  const float* ff2b = (const float*)d_in[33];
  const float* ln2g = (const float*)d_in[34];
  const float* ln2b = (const float*)d_in[35];
  const float* lnfg = (const float*)d_in[36];
  const float* lnfb = (const float*)d_in[37];
  const float* fow  = (const float*)d_in[38];
  const float* fob  = (const float*)d_in[39];
  const float* fcw  = (const float*)d_in[40];
  const float* fcb  = (const float*)d_in[41];
  const int*   ei   = (const int*)d_in[42];
  const int E  = in_sizes[42] / 2;
  const int N  = N_NODES;
  const int E2 = E + N;

  // ---- workspace carve-out ----
  char* wp = (char*)d_ws;
  size_t off = 0;
  auto alloc = [&](size_t bytes) -> void* {
    void* p = wp + off;
    off += (bytes + 255) & ~(size_t)255;
    return p;
  };
  // f16 weight copies
  h16* W1h  = (h16*)alloc((size_t)D_IN * HEADS * GAT_H * 2);
  h16* W2h  = (h16*)alloc((size_t)HEADS * GAT_H * GAT_H * 2);
  h16* fcwh = (h16*)alloc((size_t)CNN_FEAT * CNN_H * 2);
  h16* fiwh = (h16*)alloc((size_t)D_IN * TR_H * 2);
  h16* wqh  = (h16*)alloc((size_t)TR_H * TR_H * 2);
  h16* wkh  = (h16*)alloc((size_t)TR_H * TR_H * 2);
  h16* wvh  = (h16*)alloc((size_t)TR_H * TR_H * 2);
  h16* woh  = (h16*)alloc((size_t)TR_H * TR_H * 2);
  h16* ff1h = (h16*)alloc((size_t)TR_H * FF_DIM * 2);
  h16* ff2h = (h16*)alloc((size_t)FF_DIM * TR_H * 2);
  h16* foh  = (h16*)alloc((size_t)TR_H * TR_H * 2);
  h16* fcfh = (h16*)alloc((size_t)FUSED_F * OUT_F * 2);
  // f16 activations
  h16* xh     = (h16*)alloc((size_t)N * D_IN * 2);
  h16* Xh     = (h16*)alloc((size_t)N * D_IN * 2);
  h16* hreluh = (h16*)alloc((size_t)N * HEADS * GAT_H * 2);
  h16* concat = (h16*)alloc((size_t)N * CNN_FEAT * 2);
  h16* fused  = (h16*)alloc((size_t)N * FUSED_F * 2);
  h16* zh     = (h16*)alloc((size_t)N * TR_H * 2);
  h16* qh     = (h16*)alloc((size_t)N * TR_H * 2);
  h16* kh     = (h16*)alloc((size_t)N * TR_H * 2);
  h16* vh     = (h16*)alloc((size_t)N * TR_H * 2);
  h16* atth   = (h16*)alloc((size_t)N * TR_H * 2);
  h16* ffh    = (h16*)alloc((size_t)N * FF_DIM * 2);
  h16* z1h    = (h16*)alloc((size_t)N * TR_H * 2);
  h16* z3h    = (h16*)alloc((size_t)N * TR_H * 2);
  // f32 buffers
  float* h1   = (float*)alloc((size_t)N * HEADS * GAT_H * 4);
  float* as1  = (float*)alloc((size_t)N * HEADS * 4);
  float* ad1  = (float*)alloc((size_t)N * HEADS * 4);
  float* m1   = (float*)alloc((size_t)N * HEADS * 4);
  float* den1 = (float*)alloc((size_t)N * HEADS * 4);
  float* ev1  = (float*)alloc((size_t)E2 * HEADS * 4);
  float* agg1 = (float*)alloc((size_t)N * HEADS * GAT_H * 4);
  float* h2   = (float*)alloc((size_t)N * GAT_H * 4);
  float* as2  = (float*)alloc((size_t)N * 4);
  float* ad2  = (float*)alloc((size_t)N * 4);
  float* m2   = (float*)alloc((size_t)N * 4);
  float* den2 = (float*)alloc((size_t)N * 4);
  float* ev2  = (float*)alloc((size_t)E2 * 4);
  float* gat2 = (float*)alloc((size_t)N * GAT_H * 4);
  float* z    = (float*)alloc((size_t)N * TR_H * 4);
  float* ao   = (float*)alloc((size_t)N * TR_H * 4);
  float* z1   = (float*)alloc((size_t)N * TR_H * 4);
  float* ffo  = (float*)alloc((size_t)N * TR_H * 4);
  float* z2   = (float*)alloc((size_t)N * TR_H * 4);

  auto fill = [&](float* p, float v, size_t n) {
    fill_f32<<<(unsigned)((n + 255) / 256), 256, 0, stream>>>(p, v, n);
  };
  auto cvt = [&](const float* s, h16* d, size_t n) {
    f32_to_f16<<<(unsigned)((n + 255) / 256), 256, 0, stream>>>(s, d, n);
  };

  // ---- init accumulators (deterministic per call) ----
  fill(m1, -1e30f, (size_t)N * HEADS);  fill(den1, 0.0f, (size_t)N * HEADS);
  fill(agg1, 0.0f, (size_t)N * HEADS * GAT_H);
  fill(m2, -1e30f, (size_t)N);          fill(den2, 0.0f, (size_t)N);
  fill(gat2, 0.0f, (size_t)N * GAT_H);

  // ---- f16 weight/activation packing ----
  cvt(x, xh, (size_t)N * D_IN);           cvt(Xin, Xh, (size_t)N * D_IN);
  cvt(gW1, W1h, (size_t)D_IN * HEADS * GAT_H);
  cvt(gW2, W2h, (size_t)HEADS * GAT_H * GAT_H);
  cvt(cfcw, fcwh, (size_t)CNN_FEAT * CNN_H);
  cvt(fiw, fiwh, (size_t)D_IN * TR_H);
  cvt(wq, wqh, (size_t)TR_H * TR_H);  cvt(wk, wkh, (size_t)TR_H * TR_H);
  cvt(wv, wvh, (size_t)TR_H * TR_H);  cvt(wo, woh, (size_t)TR_H * TR_H);
  cvt(ff1w, ff1h, (size_t)TR_H * FF_DIM);  cvt(ff2w, ff2h, (size_t)FF_DIM * TR_H);
  cvt(fow, foh, (size_t)TR_H * TR_H);      cvt(fcw, fcfh, (size_t)FUSED_F * OUT_F);

  const int EB = 256;
  const int eth1 = E2 * HEADS;

  // ================= GAT branch =================
  // layer 1: h1 = x @ W1 (no bias pre-aggregation)
  launch_gemm(stream, xh, W1h, nullptr, h1, nullptr,
              N, HEADS * GAT_H, D_IN, D_IN, HEADS * GAT_H, HEADS * GAT_H, 0, 0, 1.0f);
  gat_attn_coef<<<(N * HEADS + EB - 1) / EB, EB, 0, stream>>>(h1, gas1, gad1, as1, ad1, N, HEADS, GAT_H);
  gat_edge_logits<<<(eth1 + EB - 1) / EB, EB, 0, stream>>>(ei, E, N, HEADS, as1, ad1, ev1, m1);
  fix_neg_inf<<<((size_t)N * HEADS + EB - 1) / EB, EB, 0, stream>>>(m1, (size_t)N * HEADS);
  gat_edge_exp<<<(eth1 + EB - 1) / EB, EB, 0, stream>>>(ei, E, N, HEADS, m1, ev1, den1);
  gat_edge_scatter<<<(eth1 + EB - 1) / EB, EB, 0, stream>>>(ei, E, N, HEADS, GAT_H, h1, ev1, den1, agg1);
  bias_act_store<<<((size_t)N * HEADS * GAT_H + EB - 1) / EB, EB, 0, stream>>>(
      agg1, gb1, hreluh, N, HEADS * GAT_H, HEADS * GAT_H, 1);
  // layer 2 (1 head): h2 = h @ W2
  launch_gemm(stream, hreluh, W2h, nullptr, h2, nullptr,
              N, GAT_H, HEADS * GAT_H, HEADS * GAT_H, GAT_H, GAT_H, 0, 0, 1.0f);
  gat_attn_coef<<<(N + EB - 1) / EB, EB, 0, stream>>>(h2, gas2, gad2, as2, ad2, N, 1, GAT_H);
  gat_edge_logits<<<(E2 + EB - 1) / EB, EB, 0, stream>>>(ei, E, N, 1, as2, ad2, ev2, m2);
  fix_neg_inf<<<(N + EB - 1) / EB, EB, 0, stream>>>(m2, (size_t)N);
  gat_edge_exp<<<(E2 + EB - 1) / EB, EB, 0, stream>>>(ei, E, N, 1, m2, ev2, den2);
  gat_edge_scatter<<<(E2 + EB - 1) / EB, EB, 0, stream>>>(ei, E, N, 1, GAT_H, h2, ev2, den2, gat2);
  bias_act_store<<<((size_t)N * GAT_H + EB - 1) / EB, EB, 0, stream>>>(
      gat2, gb2, fused /* cols 0..63 */, N, GAT_H, FUSED_F, 0);

  // ================= CNN branch =================
  cnn_conv_concat<<<(unsigned)(((size_t)N * CNN_FEAT + EB - 1) / EB), EB, 0, stream>>>(
      Xin, cw1, cb1, cw2, cb2, cw3, cb3, concat, N);
  launch_gemm(stream, concat, fcwh, cfcb, nullptr, fused + GAT_H /* cols 64..191 */,
              N, CNN_H, CNN_FEAT, CNN_FEAT, CNN_H, 0, FUSED_F, 1, 1.0f);

  // ================= Transformer branch =================
  launch_gemm(stream, Xh, fiwh, fib, z, zh, N, TR_H, D_IN, D_IN, TR_H, TR_H, TR_H, 0, 1.0f);
  launch_gemm(stream, zh, wqh, bq, nullptr, qh, N, TR_H, TR_H, TR_H, TR_H, 0, TR_H, 0, 0.125f); // 1/sqrt(HD)
  launch_gemm(stream, zh, wkh, bk, nullptr, kh, N, TR_H, TR_H, TR_H, TR_H, 0, TR_H, 0, 1.0f);
  launch_gemm(stream, zh, wvh, bv, nullptr, vh, N, TR_H, TR_H, TR_H, TR_H, 0, TR_H, 0, 1.0f);
  {
    dim3 grid(N / 16, NH);
    flash_attn_wmma<<<grid, 32, 0, stream>>>(qh, kh, vh, atth, N, TR_H);
  }
  launch_gemm(stream, atth, woh, bo, ao, nullptr, N, TR_H, TR_H, TR_H, TR_H, TR_H, 0, 0, 1.0f);
  layernorm_k<<<N, TR_H, 0, stream>>>(z, ao, ln1g, ln1b, z1, z1h, TR_H);
  launch_gemm(stream, z1h, ff1h, ff1b, nullptr, ffh, N, FF_DIM, TR_H, TR_H, FF_DIM, 0, FF_DIM, 1, 1.0f);
  launch_gemm(stream, ffh, ff2h, ff2b, ffo, nullptr, N, TR_H, FF_DIM, FF_DIM, TR_H, TR_H, 0, 0, 1.0f);
  layernorm_k<<<N, TR_H, 0, stream>>>(z1, ffo, ln2g, ln2b, z2, nullptr, TR_H);
  layernorm_k<<<N, TR_H, 0, stream>>>(z2, nullptr, lnfg, lnfb, nullptr, z3h, TR_H);
  launch_gemm(stream, z3h, foh, fob, nullptr, fused + GAT_H + CNN_H /* cols 192..319 */,
              N, TR_H, TR_H, TR_H, TR_H, 0, FUSED_F, 1, 1.0f);

  // ================= Final fusion =================
  launch_gemm(stream, fused, fcfh, fcb, (float*)d_out, nullptr,
              N, OUT_F, FUSED_F, FUSED_F, OUT_F, OUT_F, 0, 0, 1.0f);

  (void)n_in; (void)out_size; (void)ws_size;
}
